// DecoderBlock_46926812676763
// MI455X (gfx1250) — matrix operand order repaired and checked by reading, not compiled, not run on hardware
//
#include <hip/hip_runtime.h>
#include <hip/hip_bf16.h>

typedef __attribute__((ext_vector_type(16))) __bf16 v16bf;
typedef __attribute__((ext_vector_type(8)))  __bf16 bf16x8;
typedef __attribute__((ext_vector_type(8)))  float  v8f;

#define DEV __device__ __forceinline__

DEV unsigned short f2bf(float f) {
    unsigned int u = __float_as_uint(f);
    u += 0x7fffu + ((u >> 16) & 1u);       // round-to-nearest-even
    return (unsigned short)(u >> 16);
}

DEV float gelu_exact(float x) {
    return 0.5f * x * (1.0f + erff(x * 0.70710678118654752f));
}

// Build a v16bf WMMA fragment from two 16-byte-aligned 8-half chunks.
DEV v16bf ld_frag(const unsigned short* p0, const unsigned short* p1) {
    union { bf16x8 h[2]; v16bf v; } u;
    u.h[0] = *(const bf16x8*)p0;
    u.h[1] = *(const bf16x8*)p1;
    return u.v;
}

DEV v8f wmma_bf16(v16bf a, v16bf b, v8f c) {
    return __builtin_amdgcn_wmma_f32_16x16x32_bf16(false, a, false, b, (short)0, c, false, false);
}

// ---------------------------------------------------------------- f32 -> bf16
__global__ __launch_bounds__(256)
void cvt_bf16_kernel(const float* __restrict__ in, unsigned short* __restrict__ out) {
    const size_t i = ((size_t)blockIdx.x * 256u + threadIdx.x) * 4u;
    float4 f = *(const float4*)(in + i);
    ushort4 o;
    o.x = f2bf(f.x); o.y = f2bf(f.y); o.z = f2bf(f.z); o.w = f2bf(f.w);
    *(ushort4*)(out + i) = o;
}

// ---------------------------------------------------------------- LayerNorm (d=1024) -> bf16
__global__ __launch_bounds__(256)
void ln_bf16_kernel(const float* __restrict__ x, const float* __restrict__ g,
                    const float* __restrict__ be, unsigned short* __restrict__ out) {
    const int D = 1024;
    const int row = blockIdx.x;
    const int t0 = threadIdx.x * 4;
    const float* xr = x + (size_t)row * D;
    float4 v = *(const float4*)(xr + t0);
    float s  = v.x + v.y + v.z + v.w;
    float s2 = v.x*v.x + v.y*v.y + v.z*v.z + v.w*v.w;
#pragma unroll
    for (int m = 1; m < 32; m <<= 1) {
        s  += __shfl_xor(s,  m, 32);
        s2 += __shfl_xor(s2, m, 32);
    }
    __shared__ float r1[8], r2[8];
    if ((threadIdx.x & 31) == 0) { r1[threadIdx.x >> 5] = s; r2[threadIdx.x >> 5] = s2; }
    __syncthreads();
    s = 0.f; s2 = 0.f;
#pragma unroll
    for (int i = 0; i < 8; i++) { s += r1[i]; s2 += r2[i]; }
    const float mu  = s  * (1.0f / 1024.0f);
    const float var = s2 * (1.0f / 1024.0f) - mu * mu;
    const float inv = rsqrtf(var + 1e-5f);
    float4 gv = *(const float4*)(g  + t0);
    float4 bv = *(const float4*)(be + t0);
    ushort4 o;
    o.x = f2bf((v.x - mu) * inv * gv.x + bv.x);
    o.y = f2bf((v.y - mu) * inv * gv.y + bv.y);
    o.z = f2bf((v.z - mu) * inv * gv.z + bv.z);
    o.w = f2bf((v.w - mu) * inv * gv.w + bv.w);
    *(ushort4*)(out + (size_t)row * D + t0) = o;
}

// ---------------------------------------------------------------- bf16 WMMA GEMM
// C[M,N] = A[M,K] @ B[K,N] (+bias) (+res) ; A,B bf16 row-major.
// Block: 256 thr (8 waves). BM=256, BN=64, BK=32, double-buffered LDS.
// Each wave: 32x64 tile (2 row strips x 4 col tiles = 8 WMMA / K-step).
// MODE 0: out bf16 = acc+bias
// MODE 1: out f32  = res + acc + bias
// MODE 2: out bf16 = gelu(acc+bias)
template <int MODE>
__global__ __launch_bounds__(256)
void gemm_bf16_kernel(const unsigned short* __restrict__ A,
                      const unsigned short* __restrict__ Bw,
                      const float* __restrict__ bias,
                      const float* __restrict__ res,
                      void* __restrict__ outp,
                      int M, int N, int K) {
    __shared__ unsigned short ldsA[2][256 * 40];  // [m][k], stride 40 halfs (80B)
    __shared__ unsigned short ldsB[2][64 * 40];   // [n][k] transposed

    const int tid  = threadIdx.x;
    const int lane = tid & 31;
    const int w    = tid >> 5;
    const int hi   = lane >> 4;
    const int l16  = lane & 15;
    const int m0 = blockIdx.y * 256;
    const int n0 = blockIdx.x * 64;

    const v8f zero = {0.f,0.f,0.f,0.f,0.f,0.f,0.f,0.f};
    v8f acc[2][4];
#pragma unroll
    for (int rs = 0; rs < 2; rs++)
#pragma unroll
        for (int nt = 0; nt < 4; nt++) acc[rs][nt] = zero;

    // global staging pattern (exact tile coverage, no bounds needed)
    const int grow = tid >> 2, gseg = (tid & 3) * 8;   // A: rows grow+{0,64,128,192}
    const int bk   = tid >> 3, bn   = (tid & 7) * 8;   // B: one b128 per thread

    bf16x8 ra[4], rb;
    // prologue: tile 0 -> buf 0
#pragma unroll
    for (int i = 0; i < 4; i++)
        ra[i] = *(const bf16x8*)(A + (size_t)(m0 + grow + i * 64) * K + gseg);
    rb = *(const bf16x8*)(Bw + (size_t)bk * N + n0 + bn);
#pragma unroll
    for (int i = 0; i < 4; i++)
        *(bf16x8*)(&ldsA[0][(grow + i * 64) * 40 + gseg]) = ra[i];
    {
        const unsigned short* ds = (const unsigned short*)&rb;
#pragma unroll
        for (int j = 0; j < 8; j++) ldsB[0][(bn + j) * 40 + bk] = ds[j];
    }

    int cur = 0;
    for (int k0 = 0; k0 < K; k0 += 32) {
        const bool nxt = (k0 + 32) < K;
        if (nxt) {   // issue next-tile global loads before the barrier
#pragma unroll
            for (int i = 0; i < 4; i++)
                ra[i] = *(const bf16x8*)(A + (size_t)(m0 + grow + i * 64) * K + (k0 + 32) + gseg);
            rb = *(const bf16x8*)(Bw + (size_t)(k0 + 32 + bk) * N + n0 + bn);
        }
        if (k0 + 64 < K) {   // L2 prefetch two tiles ahead (global_prefetch_b8)
            __builtin_prefetch(A + (size_t)(m0 + grow) * K + (k0 + 64) + gseg, 0, 1);
            __builtin_prefetch(Bw + (size_t)(k0 + 64 + bk) * N + n0 + bn, 0, 1);
        }
        __syncthreads();                  // buf[cur] fully populated

        const unsigned short* la = ldsA[cur];
        const unsigned short* lb = ldsB[cur];
        v16bf bfr[4];
#pragma unroll
        for (int nt = 0; nt < 4; nt++) {
            const unsigned short* bp = lb + (nt * 16 + l16) * 40 + hi * 8;
            bfr[nt] = ld_frag(bp, bp + 16);
        }
#pragma unroll
        for (int rs = 0; rs < 2; rs++) {
            const unsigned short* ap = la + (w * 32 + rs * 16 + l16) * 40 + hi * 8;
            v16bf a = ld_frag(ap, ap + 16);
#pragma unroll
            for (int nt = 0; nt < 4; nt++)
                acc[rs][nt] = wmma_bf16(a, bfr[nt], acc[rs][nt]);
        }

        if (nxt) {   // stage next tile into the other buffer
            unsigned short* da = &ldsA[cur ^ 1][0];
            unsigned short* db = &ldsB[cur ^ 1][0];
#pragma unroll
            for (int i = 0; i < 4; i++)
                *(bf16x8*)(&da[(grow + i * 64) * 40 + gseg]) = ra[i];
            const unsigned short* ds = (const unsigned short*)&rb;
#pragma unroll
            for (int j = 0; j < 8; j++) db[(bn + j) * 40 + bk] = ds[j];
        }
        cur ^= 1;
    }

    // Epilogue: C layout: lane col = l16, VGPR r -> row r + 8*hi
#pragma unroll
    for (int rs = 0; rs < 2; rs++) {
#pragma unroll
        for (int nt = 0; nt < 4; nt++) {
            const int col = n0 + nt * 16 + l16;
            const float bcol = bias[col];
#pragma unroll
            for (int r = 0; r < 8; r++) {
                const int row = m0 + w * 32 + rs * 16 + r + hi * 8;
                float v = acc[rs][nt][r] + bcol;
                const size_t idx = (size_t)row * N + col;
                if (MODE == 0) {
                    ((unsigned short*)outp)[idx] = f2bf(v);
                } else if (MODE == 1) {
                    ((float*)outp)[idx] = res[idx] + v;
                } else {
                    ((unsigned short*)outp)[idx] = f2bf(gelu_exact(v));
                }
            }
        }
    }
}

// ---------------------------------------------------------------- flash attention
// qkv: [B*T, 3*D] bf16 (q|k|v); y: [B*T, D] bf16. H=16 heads, HD=64.
// Block: 128 thr (4 waves). Wave = 16 query rows; block = 64 query rows.
__global__ __launch_bounds__(128)
void attn_kernel(const unsigned short* __restrict__ qkv,
                 unsigned short* __restrict__ y) {
    const int T = 2048, D = 1024, HD = 64;
    const int RS = 3 * D;                 // qkv row stride in halfs
    __shared__ unsigned short ldsK[64 * 72];    // [key][hd]
    __shared__ unsigned short ldsVt[64 * 72];   // [hd][key]
    __shared__ unsigned short ldsP[4 * 16 * 72];// per-wave P tile [qrow][key]

    const int qb = blockIdx.x;            // query block (64 rows)
    const int h  = blockIdx.y;
    const int b  = blockIdx.z;
    const int tid = threadIdx.x, lane = tid & 31, w = tid >> 5;
    const int hi = lane >> 4, l16 = lane & 15;

    const unsigned short* Qp = qkv + (size_t)h * HD;
    const unsigned short* Kp = qkv + D     + (size_t)h * HD;
    const unsigned short* Vp = qkv + 2 * D + (size_t)h * HD;

    const int qt0 = qb * 64 + w * 16;     // wave's first query t

    // Q fragments (16 x 64, two K=32 slices), loaded once from global
    const size_t qrow = (size_t)(b * T + qt0 + l16) * RS;
    v16bf qa0 = ld_frag(Qp + qrow +      hi * 8, Qp + qrow + 16 + hi * 8);
    v16bf qa1 = ld_frag(Qp + qrow + 32 + hi * 8, Qp + qrow + 48 + hi * 8);

    const v8f zero = {0.f,0.f,0.f,0.f,0.f,0.f,0.f,0.f};
    v8f o[4];
#pragma unroll
    for (int i = 0; i < 4; i++) o[i] = zero;
    float mrow[8], lrow[8];
    const float NEG_INF = -__builtin_inff();
#pragma unroll
    for (int r = 0; r < 8; r++) { mrow[r] = NEG_INF; lrow[r] = 0.f; }
    const float scale = 0.125f;           // 1/sqrt(64)

    for (int kb = 0; kb < qb * 64 + 64; kb += 64) {
        // K tile -> ldsK [key][hd]
#pragma unroll
        for (int i = 0; i < 4; i++) {
            int cc = i * 128 + tid;
            int key = cc >> 3, seg = (cc & 7) * 8;
            bf16x8 d = *(const bf16x8*)(Kp + (size_t)(b * T + kb + key) * RS + seg);
            *(bf16x8*)(ldsK + key * 72 + seg) = d;
        }
        // V tile -> ldsVt transposed [hd][key]
#pragma unroll
        for (int i = 0; i < 4; i++) {
            int cc = i * 128 + tid;
            int key = cc >> 3, hd0 = (cc & 7) * 8;
            bf16x8 d = *(const bf16x8*)(Vp + (size_t)(b * T + kb + key) * RS + hd0);
            const unsigned short* ds = (const unsigned short*)&d;
#pragma unroll
            for (int j = 0; j < 8; j++) ldsVt[(hd0 + j) * 72 + key] = ds[j];
        }
        __syncthreads();

        // S = Q @ K^T  (16 x 64 per wave)
        v8f s[4];
#pragma unroll
        for (int nt = 0; nt < 4; nt++) {
            const unsigned short* bp = ldsK + (nt * 16 + l16) * 72 + hi * 8;
            v8f a = zero;
            a = wmma_bf16(qa0, ld_frag(bp,      bp + 16), a);
            a = wmma_bf16(qa1, ld_frag(bp + 32, bp + 48), a);
            s[nt] = a;
        }

        // mask + online softmax
#pragma unroll
        for (int r = 0; r < 8; r++) {
            const int qt = qt0 + r + hi * 8;
            float mx = NEG_INF;
#pragma unroll
            for (int nt = 0; nt < 4; nt++) {
                const int kt = kb + nt * 16 + l16;
                float v = s[nt][r] * scale;
                v = (kt <= qt) ? v : NEG_INF;
                s[nt][r] = v;
                mx = fmaxf(mx, v);
            }
#pragma unroll
            for (int m = 1; m < 16; m <<= 1) mx = fmaxf(mx, __shfl_xor(mx, m, 32));
            const float mnew = fmaxf(mrow[r], mx);
            const float alpha = __expf(mrow[r] - mnew);
            float rs = 0.f;
#pragma unroll
            for (int nt = 0; nt < 4; nt++) {
                float p = __expf(s[nt][r] - mnew);
                s[nt][r] = p;
                rs += p;
            }
#pragma unroll
            for (int m = 1; m < 16; m <<= 1) rs += __shfl_xor(rs, m, 32);
            lrow[r] = lrow[r] * alpha + rs;
            mrow[r] = mnew;
#pragma unroll
            for (int nt = 0; nt < 4; nt++) o[nt][r] *= alpha;
        }

        // stage P (bf16) to per-wave LDS for the A fragment of P@V
        unsigned short* pp = ldsP + w * 16 * 72;
#pragma unroll
        for (int r = 0; r < 8; r++)
#pragma unroll
            for (int nt = 0; nt < 4; nt++)
                pp[(r + hi * 8) * 72 + nt * 16 + l16] = f2bf(s[nt][r]);
        asm volatile("s_wait_dscnt 0" ::: "memory");

        // O += P @ V
#pragma unroll
        for (int kk = 0; kk < 64; kk += 32) {
            const unsigned short* ap = pp + l16 * 72 + kk + hi * 8;
            v16bf pa = ld_frag(ap, ap + 16);
#pragma unroll
            for (int nt = 0; nt < 4; nt++) {
                const unsigned short* bp = ldsVt + (nt * 16 + l16) * 72 + kk + hi * 8;
                o[nt] = wmma_bf16(pa, ld_frag(bp, bp + 16), o[nt]);
            }
        }
        __syncthreads();
    }

    // write y = O / l
#pragma unroll
    for (int r = 0; r < 8; r++) {
        const float inv = 1.f / lrow[r];
        const size_t row = (size_t)(b * T + qt0 + r + hi * 8);
#pragma unroll
        for (int nt = 0; nt < 4; nt++)
            y[row * D + h * HD + nt * 16 + l16] = f2bf(o[nt][r] * inv);
    }
}

// ---------------------------------------------------------------- launch
static inline size_t al256(size_t x) { return (x + 255) & ~(size_t)255; }

extern "C" void kernel_launch(void* const* d_in, const int* in_sizes, int n_in,
                              void* d_out, int out_size, void* d_ws, size_t ws_size,
                              hipStream_t stream) {
    (void)in_sizes; (void)n_in; (void)out_size; (void)ws_size;
    const int B = 4, T = 2048, D = 1024;
    const int M = B * T;                   // 8192 rows

    const float* x     = (const float*)d_in[0];
    const float* w_qkv = (const float*)d_in[1];
    const float* b_qkv = (const float*)d_in[2];
    const float* w_o   = (const float*)d_in[3];
    const float* b_o   = (const float*)d_in[4];
    const float* g1    = (const float*)d_in[5];
    const float* be1   = (const float*)d_in[6];
    const float* g2    = (const float*)d_in[7];
    const float* be2   = (const float*)d_in[8];
    const float* w1    = (const float*)d_in[9];
    const float* b1    = (const float*)d_in[10];
    const float* w2    = (const float*)d_in[11];
    const float* b2    = (const float*)d_in[12];

    // workspace layout
    char* p = (char*)d_ws;
    size_t off = 0;
    auto take = [&](size_t bytes) { char* r = p + off; off = al256(off + bytes); return r; };
    unsigned short* wqkv16 = (unsigned short*)take((size_t)D * 3 * D * 2);
    unsigned short* wo16   = (unsigned short*)take((size_t)D * D * 2);
    unsigned short* w116   = (unsigned short*)take((size_t)D * 4 * D * 2);
    unsigned short* w216   = (unsigned short*)take((size_t)4 * D * D * 2);
    unsigned short* h1     = (unsigned short*)take((size_t)M * D * 2);
    unsigned short* qkv16  = (unsigned short*)take((size_t)M * 3 * D * 2);
    unsigned short* y16    = (unsigned short*)take((size_t)M * D * 2);
    unsigned short* h2     = (unsigned short*)take((size_t)M * D * 2);
    unsigned short* mlp16  = (unsigned short*)take((size_t)M * 4 * D * 2);
    float*          x1     = (float*)take((size_t)M * D * 4);

    // 1) weights -> bf16
    cvt_bf16_kernel<<<(D * 3 * D) / 1024, 256, 0, stream>>>(w_qkv, wqkv16);
    cvt_bf16_kernel<<<(D * D) / 1024,     256, 0, stream>>>(w_o,   wo16);
    cvt_bf16_kernel<<<(D * 4 * D) / 1024, 256, 0, stream>>>(w1,    w116);
    cvt_bf16_kernel<<<(4 * D * D) / 1024, 256, 0, stream>>>(w2,    w216);

    // 2) ln1(x) -> h1 (bf16)
    ln_bf16_kernel<<<M, 256, 0, stream>>>(x, g1, be1, h1);

    // 3) qkv = h1 @ w_qkv + b_qkv  (bf16 out)
    gemm_bf16_kernel<0><<<dim3((3 * D) / 64, M / 256), 256, 0, stream>>>(
        h1, wqkv16, b_qkv, nullptr, qkv16, M, 3 * D, D);

    // 4) causal flash attention -> y16 (bf16)
    attn_kernel<<<dim3(T / 64, 16, B), 128, 0, stream>>>(qkv16, y16);

    // 5) x1 = x + y @ w_o + b_o  (f32)
    gemm_bf16_kernel<1><<<dim3(D / 64, M / 256), 256, 0, stream>>>(
        y16, wo16, b_o, x, x1, M, D, D);

    // 6) ln2(x1) -> h2 (bf16)
    ln_bf16_kernel<<<M, 256, 0, stream>>>(x1, g2, be2, h2);

    // 7) mlp = gelu(h2 @ w1 + b1) (bf16)
    gemm_bf16_kernel<2><<<dim3((4 * D) / 64, M / 256), 256, 0, stream>>>(
        h2, w116, b1, nullptr, mlp16, M, 4 * D, D);

    // 8) out = x1 + mlp @ w2 + b2 (f32)
    gemm_bf16_kernel<1><<<dim3(D / 64, M / 256), 256, 0, stream>>>(
        mlp16, w216, b2, x1, (float*)d_out, M, D, 4 * D);
}
